// TrajGRU_52372831207737
// MI455X (gfx1250) — matrix operand order repaired and checked by reading, not compiled, not run on hardware
//
#include <hip/hip_runtime.h>
#include <math.h>

typedef __attribute__((ext_vector_type(16))) _Float16 v16h;
typedef __attribute__((ext_vector_type(8)))  _Float16 v8h;
typedef __attribute__((ext_vector_type(4)))  _Float16 v4h;
typedef __attribute__((ext_vector_type(8)))  float    v8f;

namespace {
constexpr int kB   = 4, kT = 10, kH = 96, kW = 96;
constexpr int kHW  = kH * kW;          // 9216
constexpr int kP   = kB * kHW;         // 36864 GEMM rows (pixels); % 64 == 0
constexpr int kHid = 64;
constexpr int kKdef = kHid * 9;        // 576 = deform-GEMM K (k = c*9 + tap)
}

__device__ __forceinline__ float dev_lrelu(float v)   { return v >= 0.f ? v : 0.2f * v; }
__device__ __forceinline__ float dev_sigmoid(float v) { return 1.f / (1.f + __expf(-v)); }

// ---------------------------------------------------------------------------
// Weight pre-pack: f32 [O][C][kh][kw]  ->  f16 fragments in exact WMMA B lane
// layout; GEMM inner loop does ONE aligned 32-byte per-lane load.
//   wp[ ((tap*cChunks + cc)*nPad + nc)*32 + lane ][16 halves]
// lane = hi*16+lm: element e -> (n = nc*16+lm, k = cc*32 + (e & 7) + hi*8 + (e >> 3)*16)
// ---------------------------------------------------------------------------
__global__ __launch_bounds__(256) void pack_w_kernel(
    const float* __restrict__ w, _Float16* __restrict__ wp,
    int O, int C, int taps, int cChunks, int nPad)
{
  const int idx = blockIdx.x * blockDim.x + threadIdx.x;
  const int total = taps * cChunks * nPad * 32;
  if (idx >= total) return;
  const int lane = idx & 31;
  int r = idx >> 5;
  const int nc  = r % nPad;  r /= nPad;
  const int cc  = r % cChunks;
  const int tap = r / cChunks;
  const int lm = lane & 15, hi = lane >> 4;
  const int n = nc * 16 + lm;
  v16h frag;
#pragma unroll
  for (int e = 0; e < 16; ++e) {
    const int c = cc * 32 + (e & 7) + hi * 8 + (e >> 3) * 16;
    float v = 0.f;
    if (n < O && c < C) v = w[((long)n * C + c) * taps + tap];
    frag[e] = (_Float16)v;
  }
  *(v16h*)(wp + (size_t)idx * 16) = frag;
}

// ---------------------------------------------------------------------------
// Implicit-GEMM conv2d (NCHW, stride 1) via v_wmma_f32_16x16x32_f16.
// Block = 128 threads = 4 waves.  Tile: M=64 pixels x N=64 channels.
// Each wave: 4 accumulators over M-subtiles sharing ONE B fragment -> 16
// WMMAs per block per K-step.  A panel (64x32 f16) cooperatively staged in
// LDS (double buffered); next iteration's A (registers) and B (v16h) are
// prefetched before this iteration's WMMAs issue, so loads overlap compute.
// Boundary handling is clamp+mask (no divergence); epilogue transposes
// through per-wave LDS for 64B-coalesced stores.
// ---------------------------------------------------------------------------
__global__ __launch_bounds__(128) void conv2d_wmma_kernel(
    const float* __restrict__ src0, int C0, long bs0,
    const float* __restrict__ src1, long bs1,
    const _Float16* __restrict__ wp, const float* __restrict__ bias,
    float* __restrict__ out, long obs,
    int C, int O, int ks, int pad, int relu, int cChunks, int nPad)
{
  __shared__ __attribute__((aligned(16))) _Float16 aLds[2][64][40]; // pad 40 halves
  __shared__ float tLds[4][16][17];

  const int t    = threadIdx.x;
  const int wave = t >> 5;
  const int lane = t & 31;
  const int lm   = lane & 15;
  const int hi   = lane >> 4;
  const int mBase = blockIdx.x * 64;
  const int nBase = blockIdx.y * 64 + wave * 16;
  const int ncG   = blockIdx.y * 4 + wave;

  // stager role: thread t owns pixel row (t&63), 16 channels at cS
  const int mS = t & 63;
  const int cS = (t >> 6) * 16;
  const int m  = mBase + mS;
  const int b  = m / kHW;
  const int rem = m - b * kHW;
  const int y  = rem / kW;
  const int x  = rem - y * kW;
  const float* s0b = src0 + (long)b * bs0;
  const float* s1b = src1 + (long)b * bs1;

  const int taps  = ks * ks;
  const int iters = taps * cChunks;

  auto loadA = [&](int ky, int kx, int cc, float* dst) {
    const int iy = y + ky - pad;
    const int ix = x + kx - pad;
    const bool inb = (iy >= 0) & (iy < kH) & (ix >= 0) & (ix < kW);
    const int sofs = min(max(iy, 0), kH - 1) * kW + min(max(ix, 0), kW - 1);
    const int cb = cc * 32 + cS;
#pragma unroll
    for (int j = 0; j < 16; ++j) {
      const int c = cb + j;
      const bool ok = inb & (c < C);
      const int ce = min(c, C - 1);
      const float* bp = (ce < C0) ? s0b : s1b;
      const int cr = (ce < C0) ? ce : ce - C0;
      const float v = bp[(long)cr * kHW + sofs];
      dst[j] = ok ? v : 0.f;
    }
  };
  auto loadB = [&](int it) -> v16h {
    return *(const v16h*)(wp + ((size_t)(it * nPad + ncG) * 32 + lane) * 16);
  };

  v8f acc[4] = {v8f{}, v8f{}, v8f{}, v8f{}};
  float aReg[16];
  int ky = 0, kx = 0, cc = 0, p = 0;
  loadA(0, 0, 0, aReg);
  v16h bfrag = loadB(0);

  for (int it = 0; it < iters; ++it) {
    // ---- commit prefetched A panel to LDS (f32 -> f16, b64 stores) ----
#pragma unroll
    for (int q = 0; q < 4; ++q) {
      v4h st;
#pragma unroll
      for (int j = 0; j < 4; ++j) st[j] = (_Float16)aReg[q * 4 + j];
      *(v4h*)&aLds[p][mS][cS + q * 4] = st;
    }
    __syncthreads();
    // ---- prefetch next iteration (A panel regs + B fragment) ----
    int ccN = cc + 1, kyN = ky, kxN = kx;
    if (ccN == cChunks) { ccN = 0; kxN = kx + 1; if (kxN == ks) { kxN = 0; kyN = ky + 1; } }
    const bool more = (it + 1 < iters);
    float aNext[16];
    loadA(more ? kyN : ky, more ? kxN : kx, more ? ccN : cc, aNext);
    const v16h bnext = loadB(more ? it + 1 : it);
    // ---- 4 M-subtiles, one shared B fragment: 4 WMMAs ----
#pragma unroll
    for (int s = 0; s < 4; ++s) {
      const v8h a0 = *(const v8h*)&aLds[p][s * 16 + lm][hi * 8];
      const v8h a1 = *(const v8h*)&aLds[p][s * 16 + lm][16 + hi * 8];
      v16h af;
#pragma unroll
      for (int e = 0; e < 8; ++e) { af[e] = a0[e]; af[8 + e] = a1[e]; }
      acc[s] = __builtin_amdgcn_wmma_f32_16x16x32_f16(false, af, false, bfrag,
                                                      (short)0, acc[s], false, false);
    }
#pragma unroll
    for (int j = 0; j < 16; ++j) aReg[j] = aNext[j];
    bfrag = bnext; cc = ccN; ky = kyN; kx = kxN; p ^= 1;
  }

  // ---- epilogue: per-wave LDS transpose -> coalesced 64B stores ----
  // (same-wave LDS ops are in-order; no cross-wave sharing of tLds[wave])
#pragma unroll
  for (int s = 0; s < 4; ++s) {
#pragma unroll
    for (int r = 0; r < 8; ++r) tLds[wave][r + hi * 8][lm] = acc[s][r];
#pragma unroll
    for (int i = 0; i < 8; ++i) {
      const int e  = i * 32 + lane;
      const int nL = e >> 4;
      const int mL = e & 15;
      const int n  = nBase + nL;
      if (n < O) {
        const int mm = mBase + s * 16 + mL;
        const int bb = mm / kHW;
        const int ss = mm - bb * kHW;
        float v = tLds[wave][nL][mL] + bias[n];
        if (relu) v = dev_lrelu(v);
        out[(long)bb * obs + (long)n * kHW + ss] = v;
      }
    }
  }
}

// ---------------------------------------------------------------------------
// Deformable conv stage 1: bilinear sampling of `flows` at offset taps,
// producing a row-major f16 im2col matrix  sampled[m][c*9+tap]  (M x 576).
// ---------------------------------------------------------------------------
__global__ __launch_bounds__(256) void deform_sample_kernel(
    const float* __restrict__ flows, const float* __restrict__ offs,
    _Float16* __restrict__ sampled)
{
  const int idx = blockIdx.x * blockDim.x + threadIdx.x;
  if (idx >= kP * 9) return;
  const int tap = idx % 9;
  const int m = idx / 9;
  const int b = m / kHW;
  const int s = m - b * kHW;
  const int y = s / kW;
  const int x = s - y * kW;

  const float dy = offs[((long)b * 18 + tap) * kHW + s];      // off[:,0] = ch 0..8
  const float dx = offs[((long)b * 18 + 9 + tap) * kHW + s];  // off[:,1] = ch 9..17
  const float py = (float)(y + tap / 3 - 1) + dy;
  const float px = (float)(x + tap % 3 - 1) + dx;
  const float fy = floorf(py), fx = floorf(px);
  const float wy1 = py - fy, wx1 = px - fx;
  const int y0 = (int)fy, x0 = (int)fx;
  const float w00 = (1.f - wy1) * (1.f - wx1), w01 = (1.f - wy1) * wx1;
  const float w10 = wy1 * (1.f - wx1),         w11 = wy1 * wx1;
  const bool okY0 = (y0 >= 0) & (y0 < kH), okY1 = (y0 + 1 >= 0) & (y0 + 1 < kH);
  const bool okX0 = (x0 >= 0) & (x0 < kW), okX1 = (x0 + 1 >= 0) & (x0 + 1 < kW);
  const int cy0 = min(max(y0, 0), kH - 1), cy1 = min(max(y0 + 1, 0), kH - 1);
  const int cx0 = min(max(x0, 0), kW - 1), cx1 = min(max(x0 + 1, 0), kW - 1);
  const float g00 = (okY0 & okX0) ? w00 : 0.f, g01 = (okY0 & okX1) ? w01 : 0.f;
  const float g10 = (okY1 & okX0) ? w10 : 0.f, g11 = (okY1 & okX1) ? w11 : 0.f;

  const float* base = flows + (long)b * kHid * kHW;
  _Float16* orow = sampled + (long)m * kKdef + tap;
#pragma unroll 4
  for (int c = 0; c < kHid; ++c) {
    const float* f = base + (long)c * kHW;
    const float v = g00 * f[cy0 * kW + cx0] + g01 * f[cy0 * kW + cx1]
                  + g10 * f[cy1 * kW + cx0] + g11 * f[cy1 * kW + cx1];
    orow[c * 9] = (_Float16)v;
  }
}

// ---------------------------------------------------------------------------
// Deformable conv stage 2: warped = sampled(MxK f16) x def_w^T + b.
// M tile 64 (4 subtiles/wave), 4 waves tile N (64 channels).  A rows are
// contiguous f16 -> aligned global b128 loads; B pre-packed.
// ---------------------------------------------------------------------------
__global__ __launch_bounds__(128) void deform_gemm_wmma_kernel(
    const _Float16* __restrict__ A,     // [kP][576] row-major f16
    const _Float16* __restrict__ wp,    // packed (taps=1, C=576, O=64 -> nPad=4)
    const float* __restrict__ bias,
    float* __restrict__ out)            // NCHW, 64 channels
{
  __shared__ float tLds[4][16][17];
  const int t    = threadIdx.x;
  const int wave = t >> 5;
  const int lane = t & 31;
  const int lm   = lane & 15;
  const int hi   = lane >> 4;
  const int mBase = blockIdx.x * 64;
  const int nBase = wave * 16;

  v8f acc[4] = {v8f{}, v8f{}, v8f{}, v8f{}};
  constexpr int cChunks = kKdef / 32;   // 18
  for (int cc = 0; cc < cChunks; ++cc) {
    const int c0 = cc * 32;
    const v16h bfrag = *(const v16h*)(wp + ((size_t)(cc * 4 + wave) * 32 + lane) * 16);
#pragma unroll
    for (int s = 0; s < 4; ++s) {
      const _Float16* arow = A + (long)(mBase + s * 16 + lm) * kKdef + c0;
      const v8h a0 = *(const v8h*)(arow + hi * 8);
      const v8h a1 = *(const v8h*)(arow + 16 + hi * 8);
      v16h af;
#pragma unroll
      for (int e = 0; e < 8; ++e) { af[e] = a0[e]; af[8 + e] = a1[e]; }
      acc[s] = __builtin_amdgcn_wmma_f32_16x16x32_f16(false, af, false, bfrag,
                                                      (short)0, acc[s], false, false);
    }
  }
#pragma unroll
  for (int s = 0; s < 4; ++s) {
#pragma unroll
    for (int r = 0; r < 8; ++r) tLds[wave][r + hi * 8][lm] = acc[s][r];
#pragma unroll
    for (int i = 0; i < 8; ++i) {
      const int e  = i * 32 + lane;
      const int nL = e >> 4;
      const int mL = e & 15;
      const int n  = nBase + nL;
      const int mm = mBase + s * 16 + mL;
      const int bb = mm / kHW;
      const int ss = mm - bb * kHW;
      out[((long)bb * kHid + n) * kHW + ss] = tLds[wave][nL][mL] + bias[n];
    }
  }
}

// ---------------------------------------------------------------------------
// Fused GRU gating: z,r = sigmoid(split(comb)); h_new = lrelu(ari + r*brh);
// h = (1-z)*h_new + z*h_prev
// ---------------------------------------------------------------------------
__global__ __launch_bounds__(256) void gru_fuse_kernel(
    const float* __restrict__ comb, const float* __restrict__ ari,
    const float* __restrict__ brh, const float* __restrict__ hprev, long hbs,
    float* __restrict__ hout, long obs)
{
  const int idx = blockIdx.x * blockDim.x + threadIdx.x;
  if (idx >= kP * kHid) return;
  const int b  = idx / (kHid * kHW);
  const int r0 = idx - b * kHid * kHW;
  const int c  = r0 / kHW;
  const int s  = r0 - c * kHW;
  const float z = dev_sigmoid(comb[((long)b * 128 + c) * kHW + s]);
  const float r = dev_sigmoid(comb[((long)b * 128 + 64 + c) * kHW + s]);
  const float hn = dev_lrelu(ari[idx] + r * brh[idx]);
  const float hp = hprev[(long)b * hbs + (long)c * kHW + s];
  hout[(long)b * obs + (long)c * kHW + s] = (1.f - z) * hn + z * hp;
}

__global__ void zero_f32_kernel(float* __restrict__ p, long n) {
  const long i = (long)blockIdx.x * blockDim.x + threadIdx.x;
  if (i < n) p[i] = 0.f;
}

// ---------------------------------------------------------------------------
static inline char* ws_bump(char*& cur, size_t bytes) {
  char* r = cur;
  cur += (bytes + 255) & ~(size_t)255;
  return r;
}

extern "C" void kernel_launch(void* const* d_in, const int* in_sizes, int n_in,
                              void* d_out, int out_size, void* d_ws, size_t ws_size,
                              hipStream_t stream)
{
  (void)in_sizes; (void)n_in; (void)out_size; (void)ws_size;
  // setup_inputs() dict order: x, then l0 params (insertion order), then l1.
  const float* x = (const float*)d_in[0];
  const float* prm[2][14];
  for (int l = 0; l < 2; ++l)
    for (int i = 0; i < 14; ++i)
      prm[l][i] = (const float*)d_in[1 + l * 14 + i];
  enum { UV1W, UV1B, UV2W, UV2B, OFFW, OFFB, DEFW, DEFB, CNVW, CNVB, RIW, RIB, RHW, RHB };

  // ---------------- workspace ----------------
  char* cur = (char*)d_ws;
  float*    seq0   = (float*)ws_bump(cur, (size_t)kT * kP * kHid * 4);  // layer-0 outputs [B,T,64,H,W]
  float*    h0     = (float*)ws_bump(cur, (size_t)kP * kHid * 4);
  float*    fbuf   = (float*)ws_bump(cur, (size_t)kP * 32 * 4);
  float*    flows  = (float*)ws_bump(cur, (size_t)kP * kHid * 4);
  float*    offs   = (float*)ws_bump(cur, (size_t)kP * 18 * 4);
  float*    warped = (float*)ws_bump(cur, (size_t)kP * kHid * 4);
  float*    comb   = (float*)ws_bump(cur, (size_t)kP * 128 * 4);
  float*    ari    = (float*)ws_bump(cur, (size_t)kP * kHid * 4);
  float*    brh    = (float*)ws_bump(cur, (size_t)kP * kHid * 4);
  _Float16* smp    = (_Float16*)ws_bump(cur, (size_t)kP * kKdef * 2);

  // ---------------- weight packing (once per call; reused across T) -------
  // conv order: 0=uv1 1=uv2 2=off 3=def(im2col GEMM) 4=conv 5=ri 6=rh
  const int wIdx[7] = {UV1W, UV2W, OFFW, DEFW, CNVW, RIW, RHW};
  const int oDim[7] = {32, 64, 18, 64, 128, 64, 64};
  const int kDim[7] = {5, 5, 3, 1, 3, 3, 3};   // def packed as taps=1, C=576
  _Float16* wpk[2][7];
  int cChk[2][7], nPad[2][7];
  for (int l = 0; l < 2; ++l) {
    const int inC = l ? kHid : 8;
    const int cDim[7] = {inC + kHid, 32, kHid, kKdef, inC + kHid, inC, kHid};
    for (int j = 0; j < 7; ++j) {
      const int taps = kDim[j] * kDim[j];
      const int cc = (cDim[j] + 31) / 32;
      const int np = ((oDim[j] + 63) / 64) * 4;
      cChk[l][j] = cc; nPad[l][j] = np;
      wpk[l][j] = (_Float16*)ws_bump(cur, (size_t)taps * cc * np * 512 * 2);
      const int total = taps * cc * np * 32;
      pack_w_kernel<<<(total + 255) / 256, 256, 0, stream>>>(
          prm[l][wIdx[j]], wpk[l][j], oDim[j], cDim[j], taps, cc, np);
    }
  }

  zero_f32_kernel<<<(kP * kHid + 255) / 256, 256, 0, stream>>>(h0, (long)kP * kHid);

  const dim3 blk(128);
  const int gx = kP / 64;                      // 576 M-tiles

  for (int l = 0; l < 2; ++l) {
    const float* const* p = prm[l];
    const int inC = (l == 0) ? 8 : kHid;
    float* seq = (l == 0) ? seq0 : (float*)d_out;
    const float* src = (l == 0) ? x : seq0;
    const long sbs = (long)kT * inC * kHW;

    for (int t = 0; t < kT; ++t) {
      const float* xin = src + (long)t * inC * kHW;
      const float* hprev; long hbs;
      if (t == 0) { hprev = h0;                               hbs = (long)kHid * kHW; }
      else        { hprev = seq + (long)(t - 1) * kHid * kHW; hbs = (long)kT * kHid * kHW; }
      float* hout = seq + (long)t * kHid * kHW;
      const long obs_h = (long)kT * kHid * kHW;

      // f = lrelu(conv5x5(concat(x,h)))        O=32
      conv2d_wmma_kernel<<<dim3(gx, nPad[l][0] / 4), blk, 0, stream>>>(
          xin, inC, sbs, hprev, hbs, wpk[l][0], p[UV1B],
          fbuf, 32L * kHW, inC + kHid, 32, 5, 2, 1, cChk[l][0], nPad[l][0]);
      // flows = conv5x5(f)                     O=64
      conv2d_wmma_kernel<<<dim3(gx, nPad[l][1] / 4), blk, 0, stream>>>(
          fbuf, 32, 32L * kHW, fbuf, 0, wpk[l][1], p[UV2B],
          flows, 64L * kHW, 32, 64, 5, 2, 0, cChk[l][1], nPad[l][1]);
      // offs = conv3x3(flows)                  O=18
      conv2d_wmma_kernel<<<dim3(gx, nPad[l][2] / 4), blk, 0, stream>>>(
          flows, kHid, 64L * kHW, flows, 0, wpk[l][2], p[OFFB],
          offs, 18L * kHW, kHid, 18, 3, 1, 0, cChk[l][2], nPad[l][2]);
      // warped = deform_conv(flows, offs)
      deform_sample_kernel<<<(kP * 9 + 255) / 256, 256, 0, stream>>>(flows, offs, smp);
      deform_gemm_wmma_kernel<<<dim3(gx), blk, 0, stream>>>(smp, wpk[l][3], p[DEFB], warped);
      // comb = conv3x3(concat(x,warped))       O=128
      conv2d_wmma_kernel<<<dim3(gx, nPad[l][4] / 4), blk, 0, stream>>>(
          xin, inC, sbs, warped, 64L * kHW, wpk[l][4], p[CNVB],
          comb, 128L * kHW, inC + kHid, 128, 3, 1, 0, cChk[l][4], nPad[l][4]);
      // ari = conv3x3(x)                       O=64
      conv2d_wmma_kernel<<<dim3(gx, nPad[l][5] / 4), blk, 0, stream>>>(
          xin, inC, sbs, xin, 0, wpk[l][5], p[RIB],
          ari, 64L * kHW, inC, 64, 3, 1, 0, cChk[l][5], nPad[l][5]);
      // brh = conv3x3(warped)                  O=64
      conv2d_wmma_kernel<<<dim3(gx, nPad[l][6] / 4), blk, 0, stream>>>(
          warped, kHid, 64L * kHW, warped, 0, wpk[l][6], p[RHB],
          brh, 64L * kHW, kHid, 64, 3, 1, 0, cChk[l][6], nPad[l][6]);
      // gates + state update -> hout (inside seq / d_out)
      gru_fuse_kernel<<<(kP * kHid + 255) / 256, 256, 0, stream>>>(
          comb, ari, brh, hprev, hbs, hout, obs_h);
    }
  }
}
